// RNN_17806934409416
// MI455X (gfx1250) — compile-verified, matrix operand order checked by reading, not run on hardware
//
#include <hip/hip_runtime.h>
#include <hip/hip_bf16.h>
#include <stdint.h>

#define EMBD 1024
#define HIDD 1024
#define BATCH 64
#define SEQ 512
#define MROWS (BATCH * SEQ)

typedef __attribute__((ext_vector_type(16))) __bf16 v16bf;
typedef __attribute__((ext_vector_type(8)))  __bf16 v8bf;
typedef __attribute__((ext_vector_type(8)))  float  v8f;
typedef __attribute__((ext_vector_type(4)))  unsigned int v4u;
typedef __attribute__((ext_vector_type(8)))  int          v8i;
typedef __attribute__((ext_vector_type(4)))  int          v4i;

union ABf {
    v16bf v;
    v8bf  h[2];
};

// ---------------------------------------------------------------------------
// Kernel 1: fp32 -> bf16 conversion of embeds / W_ih / W_hh, plus zeroing of
// the h double-buffer and the per-step sync counters (must be re-zeroed every
// launch for determinism under graph replay).
// ---------------------------------------------------------------------------
__global__ void rnn_init_kernel(const float* __restrict__ emb,
                                const float* __restrict__ wih,
                                const float* __restrict__ whh,
                                __bf16* __restrict__ embB,
                                __bf16* __restrict__ wihB,
                                __bf16* __restrict__ whhB,
                                unsigned int* __restrict__ hbuf32,
                                unsigned int* __restrict__ cnt)
{
    size_t gid    = (size_t)blockIdx.x * blockDim.x + threadIdx.x;
    size_t stride = (size_t)gridDim.x * blockDim.x;

    for (size_t i = gid; i < (size_t)MROWS * EMBD; i += stride)
        embB[i] = (__bf16)emb[i];
    for (size_t i = gid; i < (size_t)HIDD * EMBD; i += stride) {
        wihB[i] = (__bf16)wih[i];
        whhB[i] = (__bf16)whh[i];
    }
    // 2 * 64 * 1024 bf16 = 65536 dwords
    for (size_t i = gid; i < (size_t)2 * BATCH * HIDD / 2; i += stride)
        hbuf32[i] = 0u;
    for (size_t i = gid; i < (size_t)4 * SEQ; i += stride)
        cnt[i] = 0u;
}

// ---------------------------------------------------------------------------
// Kernel 2: x_proj[m, n] = sum_e embeds[m, e] * W_ih[n, e] + b_ih[n] + b_hh[n]
// M = 32768, N = 1024, K = 1024.  Each wave computes a 16M x 64N tile
// (4 accumulators), K-loop of 32 x v_wmma_f32_16x16x32_bf16 per tile.
// ---------------------------------------------------------------------------
__global__ void __launch_bounds__(256)
xproj_wmma_kernel(const __bf16* __restrict__ A,    // (MROWS, EMBD) bf16
                  const __bf16* __restrict__ Wb,   // (HIDD, EMBD)  bf16
                  const float*  __restrict__ b_ih,
                  const float*  __restrict__ b_hh,
                  float*        __restrict__ xp)   // (MROWS, HIDD) f32
{
    const int wid   = (int)((blockIdx.x * blockDim.x + threadIdx.x) >> 5);
    const int lane  = (int)(threadIdx.x & 31u);
    const int mtile = wid >> 4;   // 0..2047
    const int ng    = wid & 15;   // 64 output cols per wave
    const int r     = lane & 15;
    const int hi    = lane >> 4;

    const __bf16* arow = A + (size_t)(mtile * 16 + r) * EMBD + 8 * hi;
    const __bf16* brow = Wb + (size_t)(ng * 64 + r) * EMBD + 8 * hi;

    v8f acc[4];
    acc[0] = {}; acc[1] = {}; acc[2] = {}; acc[3] = {};

    for (int k = 0; k < EMBD; k += 32) {
        ABf a;
        a.h[0] = *(const v8bf*)(arow + k);
        a.h[1] = *(const v8bf*)(arow + k + 16);
#pragma unroll
        for (int j = 0; j < 4; ++j) {
            ABf b;
            const __bf16* bj = brow + (size_t)j * 16 * EMBD;
            b.h[0] = *(const v8bf*)(bj + k);
            b.h[1] = *(const v8bf*)(bj + k + 16);
            acc[j] = __builtin_amdgcn_wmma_f32_16x16x32_bf16(
                false, a.v, false, b.v, (short)0, acc[j], false, false);
        }
    }

#pragma unroll
    for (int j = 0; j < 4; ++j) {
        const int n    = ng * 64 + j * 16 + r;     // lane -> N column
        const float bs = b_ih[n] + b_hh[n];
#pragma unroll
        for (int r8 = 0; r8 < 8; ++r8) {
            const int m = mtile * 16 + r8 + 8 * hi; // C/D layout: M = r8 + 8*hi
            xp[(size_t)m * HIDD + n] = acc[j][r8] + bs;
        }
    }
}

// ---------------------------------------------------------------------------
// Kernel 3: recurrence.  Grid = 4 batch-groups x 16 H-groups = 64 resident
// workgroups of 128 threads (4 waves, one 16x16 N-tile each).  Per step:
//   - spin (acquire) on the previous step's counter of this batch-group
//   - TDM: tensor_load_to_lds stages h_{t-1} (16 x 1024 bf16, 32 KB) into LDS
//   - 32 x v_wmma_f32_16x16x32_bf16 over K=1024 (A from LDS, B = W_hh rows)
//   - epilogue: + x_proj, tanh, store hs (f32) and h_t slice (bf16, double
//     buffered), fence, release-increment the step counter.
// ---------------------------------------------------------------------------
__global__ void __launch_bounds__(128)
rnn_recurrence_kernel(const __bf16* __restrict__ Whh,  // (HIDD, HIDD) bf16
                      const float*  __restrict__ xp,   // (MROWS, HIDD) f32
                      float*        __restrict__ hs,   // (SEQ, BATCH, HIDD) f32
                      __bf16*       hbuf,              // (2, BATCH, HIDD) bf16
                      unsigned int* cnt)               // (4, SEQ)
{
    __shared__ __bf16 hA[16 * HIDD];   // only LDS object -> LDS byte offset 0

    const int bg   = (int)(blockIdx.x >> 4);  // batch group, 16 batches
    const int hg   = (int)(blockIdx.x & 15);  // 64 H channels
    const int w    = (int)(threadIdx.x >> 5);
    const int lane = (int)(threadIdx.x & 31u);
    const int r    = lane & 15;
    const int hi   = lane >> 4;
    const int n    = hg * 64 + w * 16 + r;    // this lane's N column

    const __bf16* wrow = Whh + (size_t)n * HIDD + 8 * hi;

    for (int t = 0; t < SEQ; ++t) {
        v8f acc = {};

        if (t > 0) {
            if (threadIdx.x == 0) {
                while (__hip_atomic_load(&cnt[bg * SEQ + (t - 1)],
                                         __ATOMIC_ACQUIRE,
                                         __HIP_MEMORY_SCOPE_AGENT) < 16u) {
                    __builtin_amdgcn_s_sleep(2);
                }
            }
            __syncthreads();

            if (threadIdx.x < 32) {
                // --- Tensor Data Mover: h_{t-1} tile -> LDS -------------
                // D# group0: count=1, lds_addr=0, global_addr, type=2
                uint64_t ga = (uint64_t)(uintptr_t)(hbuf
                              + (size_t)((t - 1) & 1) * BATCH * HIDD
                              + (size_t)bg * 16 * HIDD);
                v4u g0 = { 1u, 0u, (unsigned int)ga,
                           ((unsigned int)(ga >> 32) & 0x01FFFFFFu) | 0x80000000u };
                // D# group1: data_size=2B, tensor 1024 x 16, tile 1024 x 16,
                //            dim0 stride = 1024 elements
                v8i g1 = { (int)0x00010000,        // data_size=1 (2 bytes)
                           (int)(HIDD << 16),      // tensor_dim0[15:0]
                           (int)(16 << 16),        // tensor_dim1[15:0]
                           (int)(HIDD << 16),      // tile_dim0
                           16,                     // tile_dim1
                           HIDD,                   // tensor_dim0_stride[31:0]
                           0, 0 };
                v4i gz4 = { 0, 0, 0, 0 };
                v8i gz8 = { 0, 0, 0, 0, 0, 0, 0, 0 };
                // amdgpu-toolchain (clang-23) 6-arg form:
                // (v4u g0, v8i g1, v4i, v4i, v8i, i32 cpol)
                __builtin_amdgcn_tensor_load_to_lds(g0, g1, gz4, gz4, gz8, 0);
                __builtin_amdgcn_s_wait_tensorcnt(0);
            }
            __syncthreads();

            // The TDM write to LDS is invisible to the compiler (the builtin
            // doesn't reference hA).  Escape hA's address + memory clobber so
            // the ds_load_b128 reads below cannot be folded to undef.
            asm volatile("" : : "r"(&hA[0]) : "memory");

            const __bf16* arow = hA + r * HIDD + 8 * hi;  // ds_load_b128 path
            for (int k = 0; k < HIDD; k += 32) {
                // prefetch next K-block of the W_hh row (global_prefetch_b8)
                if (k + 32 < HIDD)
                    __builtin_prefetch((const void*)(wrow + k + 32), 0, 3);
                ABf a, b;
                a.h[0] = *(const v8bf*)(arow + k);
                a.h[1] = *(const v8bf*)(arow + k + 16);
                b.h[0] = *(const v8bf*)(wrow + k);
                b.h[1] = *(const v8bf*)(wrow + k + 16);
                acc = __builtin_amdgcn_wmma_f32_16x16x32_bf16(
                    false, a.v, false, b.v, (short)0, acc, false, false);
            }
        }

        // Epilogue: bias'd input projection + tanh; write hs and h double buf
#pragma unroll
        for (int r8 = 0; r8 < 8; ++r8) {
            const int b_ = bg * 16 + r8 + 8 * hi;           // batch index
            float v = acc[r8] + xp[((size_t)b_ * SEQ + t) * HIDD + n];
            v = tanhf(v);
            hs[((size_t)t * BATCH + b_) * HIDD + n] = v;
            hbuf[(size_t)(t & 1) * BATCH * HIDD + (size_t)b_ * HIDD + n] = (__bf16)v;
        }

        __threadfence();
        __syncthreads();
        if (threadIdx.x == 0) {
            __hip_atomic_fetch_add(&cnt[bg * SEQ + t], 1u,
                                   __ATOMIC_RELEASE, __HIP_MEMORY_SCOPE_AGENT);
        }
    }
}

// ---------------------------------------------------------------------------
extern "C" void kernel_launch(void* const* d_in, const int* in_sizes, int n_in,
                              void* d_out, int out_size, void* d_ws, size_t ws_size,
                              hipStream_t stream) {
    const float* emb  = (const float*)d_in[0];
    const float* wih  = (const float*)d_in[1];
    const float* whh  = (const float*)d_in[2];
    const float* b_ih = (const float*)d_in[3];
    const float* b_hh = (const float*)d_in[4];
    float* hs = (float*)d_out;

    // Workspace layout (all 256B-aligned): ~206 MB total
    char* ws = (char*)d_ws;
    float*  xpj   = (float*)ws;   ws += (size_t)MROWS * HIDD * sizeof(float);   // 128 MB
    __bf16* embB  = (__bf16*)ws;  ws += (size_t)MROWS * EMBD * sizeof(__bf16);  //  64 MB
    __bf16* wihB  = (__bf16*)ws;  ws += (size_t)HIDD * EMBD * sizeof(__bf16);   //   2 MB
    __bf16* whhB  = (__bf16*)ws;  ws += (size_t)HIDD * HIDD * sizeof(__bf16);   //   2 MB
    __bf16* hbuf  = (__bf16*)ws;  ws += (size_t)2 * BATCH * HIDD * sizeof(__bf16);
    unsigned int* cnt = (unsigned int*)ws;

    rnn_init_kernel<<<4096, 256, 0, stream>>>(emb, wih, whh, embB, wihB, whhB,
                                              (unsigned int*)hbuf, cnt);

    // 32768 waves: 2048 M-tiles x 16 N-groups; 8 waves per block
    xproj_wmma_kernel<<<4096, 256, 0, stream>>>(embB, wihB, b_ih, b_hh, xpj);

    // 4 batch-groups x 16 H-groups = 64 persistent workgroups
    rnn_recurrence_kernel<<<64, 128, 0, stream>>>(whhB, xpj, hs, hbuf, cnt);
}
